// MSARowAttentionWithPairBias_55688545960634
// MI455X (gfx1250) — matrix operand order repaired and checked by reading, not compiled, not run on hardware
//
#include <hip/hip_runtime.h>
#include <hip/hip_bf16.h>
#include <stdint.h>

typedef __attribute__((ext_vector_type(16))) __bf16 v16bf;
typedef __attribute__((ext_vector_type(8)))  float  v8f;

#define CM 256
#define CZ 128
#define CH 32
#define NH 8
#define SEQ 256
#define LLEN 256
#define ROWS (SEQ*LLEN)   /* 65536 */

union Frag16 { v16bf v; uint4 q[2]; };

__device__ __forceinline__ float bf2f(unsigned short u) {
    union { unsigned int i; float f; } c; c.i = ((unsigned int)u) << 16; return c.f;
}
__device__ __forceinline__ unsigned short f2bf(float f) {
    union { float f; unsigned int i; } c; c.f = f;
    unsigned int r = c.i + 0x7FFFu + ((c.i >> 16) & 1u);
    return (unsigned short)(r >> 16);
}
__device__ __forceinline__ v8f wmma_bf16(const Frag16& a, const Frag16& b, v8f c) {
    return __builtin_amdgcn_wmma_f32_16x16x32_bf16(false, a.v, false, b.v,
                                                   (short)0, c, false, false);
}

// ---------------- LayerNorm of m -> bf16 ----------------
__global__ __launch_bounds__(256)
void ln_m_kernel(const float* __restrict__ m, const float* __restrict__ g,
                 const float* __restrict__ b, unsigned short* __restrict__ out) {
    int wave = threadIdx.x >> 5, lane = threadIdx.x & 31;
    int row = blockIdx.x * 8 + wave;
    const float* src = m + (size_t)row * CM;
    float v[8]; float s = 0.f, ss = 0.f;
#pragma unroll
    for (int i = 0; i < 8; ++i) { v[i] = src[lane + 32*i]; s += v[i]; ss += v[i]*v[i]; }
#pragma unroll
    for (int msk = 1; msk < 32; msk <<= 1) { s += __shfl_xor(s, msk); ss += __shfl_xor(ss, msk); }
    float mu = s * (1.f/256.f);
    float var = ss * (1.f/256.f) - mu*mu;
    float rs = rsqrtf(var + 1e-5f);
    unsigned short* dst = out + (size_t)row * CM;
#pragma unroll
    for (int i = 0; i < 8; ++i) {
        int c = lane + 32*i;
        dst[c] = f2bf((v[i]-mu)*rs*g[c] + b[c]);
    }
}

// ---------------- LayerNorm of z fused with @Wb -> bias[h][i][j] fp32 ----------------
__global__ __launch_bounds__(256)
void pair_bias_kernel(const float* __restrict__ z, const float* __restrict__ g,
                      const float* __restrict__ b, const float* __restrict__ Wb,
                      float* __restrict__ biasout) {
    int wave = threadIdx.x >> 5, lane = threadIdx.x & 31;
    int row = blockIdx.x * 8 + wave;   // row = i*256 + j
    const float* src = z + (size_t)row * CZ;
    float v[4]; float s = 0.f, ss = 0.f;
#pragma unroll
    for (int i = 0; i < 4; ++i) { v[i] = src[lane + 32*i]; s += v[i]; ss += v[i]*v[i]; }
#pragma unroll
    for (int msk = 1; msk < 32; msk <<= 1) { s += __shfl_xor(s, msk); ss += __shfl_xor(ss, msk); }
    float mu = s * (1.f/128.f);
    float var = ss * (1.f/128.f) - mu*mu;
    float rs = rsqrtf(var + 1e-5f);
    float acc[NH];
#pragma unroll
    for (int h = 0; h < NH; ++h) acc[h] = 0.f;
#pragma unroll
    for (int i = 0; i < 4; ++i) {
        int c = lane + 32*i;
        float zn = (v[i]-mu)*rs*g[c] + b[c];
#pragma unroll
        for (int h = 0; h < NH; ++h) acc[h] += zn * Wb[c*NH + h];
    }
#pragma unroll
    for (int msk = 1; msk < 32; msk <<= 1) {
#pragma unroll
        for (int h = 0; h < NH; ++h) acc[h] += __shfl_xor(acc[h], msk);
    }
    if (lane == 0) {
#pragma unroll
        for (int h = 0; h < NH; ++h) biasout[(size_t)h*ROWS + row] = acc[h];
    }
}

// ---------------- transpose + fp32->bf16 weight convert: Wt[n][k] = W[k][n] ----------------
__global__ void wt_convert_kernel(const float* __restrict__ W, unsigned short* __restrict__ Wt,
                                  int K, int N) {
    int k = blockIdx.x * 16 + threadIdx.x;
    int n = blockIdx.y * 16 + threadIdx.y;
    if (k < K && n < N) Wt[(size_t)n*K + k] = f2bf(W[(size_t)k*N + n]);
}

// ---------------- fused Q/K/V/gate projection: A loaded once, 4 WMMAs per k-step ----------------
__global__ __launch_bounds__(256)
void proj_fused_kernel(const unsigned short* __restrict__ A,
                       const unsigned short* __restrict__ WtQ,
                       const unsigned short* __restrict__ WtK,
                       const unsigned short* __restrict__ WtV,
                       const unsigned short* __restrict__ WtG,
                       const float* __restrict__ bg,
                       unsigned short* __restrict__ qp,
                       unsigned short* __restrict__ kp,
                       unsigned short* __restrict__ vp,
                       unsigned short* __restrict__ gatep) {
    int wave = threadIdx.x >> 5, lane = threadIdx.x & 31;
    int wid = blockIdx.x * 8 + wave;
    int n0 = (wid & 15) << 4;
    int m0 = (wid >> 4) << 4;
    int hi = lane >> 4, lo = lane & 15;

    v8f aq = {}, ak = {}, av = {}, ag = {};
    const unsigned short* arow = A + (size_t)(m0 + lo) * 256;
    size_t boff = (size_t)(n0 + lo) * 256;
#pragma unroll
    for (int kk = 0; kk < 8; ++kk) {
        int kb = kk * 32;
        Frag16 af, fq, fk, fv, fg;
        // A 16x32 bf16 fragment: lane row = m0+lo; K = kb+16*hi..+15 contiguous
        af.q[0] = *(const uint4*)(arow + kb + 16*hi);
        af.q[1] = *(const uint4*)(arow + kb + 16*hi + 8);
        // B 32x16 fragments: lane col = n0+lo; K = kb+16*hi..+15 contiguous
        const unsigned short* pq = WtQ + boff + kb + 16*hi;
        const unsigned short* pk = WtK + boff + kb + 16*hi;
        const unsigned short* pv = WtV + boff + kb + 16*hi;
        const unsigned short* pg = WtG + boff + kb + 16*hi;
        fq.q[0] = *(const uint4*)(pq); fq.q[1] = *(const uint4*)(pq + 8);
        fk.q[0] = *(const uint4*)(pk); fk.q[1] = *(const uint4*)(pk + 8);
        fv.q[0] = *(const uint4*)(pv); fv.q[1] = *(const uint4*)(pv + 8);
        fg.q[0] = *(const uint4*)(pg); fg.q[1] = *(const uint4*)(pg + 8);
        aq = wmma_bf16(af, fq, aq);
        ak = wmma_bf16(af, fk, ak);
        av = wmma_bf16(af, fv, av);
        ag = wmma_bf16(af, fg, ag);
    }
    float bgn = bg[n0 + lo];
#pragma unroll
    for (int j = 0; j < 8; ++j) {
        int mr = m0 + j + 8*hi;          // D layout: row = j + 8*(lane>=16)
        int n  = n0 + lo;                // col = lane & 15
        int s = mr >> 8, l = mr & 255;
        int h = n >> 5,  c = n & 31;
        size_t pidx = (((size_t)l*NH + h)*SEQ + s)*CH + c;   // [l][h][s][c]
        qp[pidx] = f2bf(aq[j]);
        kp[pidx] = f2bf(ak[j]);
        vp[pidx] = f2bf(av[j]);
        float x = ag[j] + bgn;
        gatep[(size_t)mr*256 + n] = f2bf(1.f / (1.f + __expf(-x)));
    }
}

// ---------------- output GEMM: out[m][n] = (g*o)[m][:] . WtO[n][:] + bo[n], fp32 ----------------
__global__ __launch_bounds__(256)
void gemm_out_kernel(const unsigned short* __restrict__ A,
                     const unsigned short* __restrict__ Wt,
                     const float* __restrict__ bias,
                     float* __restrict__ dst) {
    int wave = threadIdx.x >> 5, lane = threadIdx.x & 31;
    int wid = blockIdx.x * 8 + wave;
    int n0 = (wid & 15) << 4;
    int m0 = (wid >> 4) << 4;
    int hi = lane >> 4, lo = lane & 15;

    v8f acc = {};
    const unsigned short* arow = A  + (size_t)(m0 + lo) * 256;
    const unsigned short* bcol = Wt + (size_t)(n0 + lo) * 256;
#pragma unroll
    for (int kk = 0; kk < 8; ++kk) {
        int kb = kk * 32;
        Frag16 af, bf;
        af.q[0] = *(const uint4*)(arow + kb + 16*hi);
        af.q[1] = *(const uint4*)(arow + kb + 16*hi + 8);
        bf.q[0] = *(const uint4*)(bcol + kb + 16*hi);
        bf.q[1] = *(const uint4*)(bcol + kb + 16*hi + 8);
        acc = wmma_bf16(af, bf, acc);
    }
    float bn = bias[n0 + lo];
#pragma unroll
    for (int j = 0; j < 8; ++j) {
        int mr = m0 + j + 8*hi;
        dst[(size_t)mr*256 + n0 + lo] = acc[j] + bn;
    }
}

// ---------------- flash attention per (l,h): softmax(QK^T/sqrt(c)+bias) V, gated ----------------
__global__ __launch_bounds__(512)
void attn_kernel(const unsigned short* __restrict__ qp,
                 const unsigned short* __restrict__ kp,
                 const unsigned short* __restrict__ vp,
                 const float* __restrict__ biasp,
                 const unsigned short* __restrict__ gate,
                 unsigned short* __restrict__ go) {
    __shared__ unsigned short Ksm[SEQ*CH];      // [t][c]   16 KB
    __shared__ unsigned short Vt[CH*SEQ];       // [c][t]   16 KB
    __shared__ unsigned short Pscr[16][16*32];  // per-wave P bounce, 16 KB

    int lh = blockIdx.x;                // l*8 + h
    int l = lh >> 3, h = lh & 7;
    int tid = threadIdx.x;
    size_t base = (size_t)lh * SEQ * CH;

    for (int idx = tid; idx < SEQ*CH; idx += 512) {
        int t = idx >> 5, c = idx & 31;
        Ksm[idx] = kp[base + idx];
        Vt[c*SEQ + t] = vp[base + idx];
    }
    __syncthreads();

    int wave = tid >> 5, lane = tid & 31;
    int hi = lane >> 4, lo = lane & 15;
    int srow0 = wave * 16;

    Frag16 qf;
    const unsigned short* qrow = qp + base + (size_t)(srow0 + lo) * CH;
    qf.q[0] = *(const uint4*)(qrow + 16*hi);
    qf.q[1] = *(const uint4*)(qrow + 16*hi + 8);

    float mrow[8], lrow[8];
    v8f o0 = {}, o1 = {};
#pragma unroll
    for (int j = 0; j < 8; ++j) { mrow[j] = -3.0e38f; lrow[j] = 0.f; }

    const float scale = 0.17677669529663687f;   // 1/sqrt(32)
    const float* brow = biasp + (size_t)h * ROWS;

    for (int t0 = 0; t0 < SEQ; t0 += 32) {
        // logits tile [16 x 32]: two WMMAs; B = K^T from LDS (contiguous per lane)
        Frag16 kf0, kf1;
        const unsigned short* k0 = Ksm + (t0 + lo) * CH;
        const unsigned short* k1 = Ksm + (t0 + 16 + lo) * CH;
        kf0.q[0] = *(const uint4*)(k0 + 16*hi);  kf0.q[1] = *(const uint4*)(k0 + 16*hi + 8);
        kf1.q[0] = *(const uint4*)(k1 + 16*hi);  kf1.q[1] = *(const uint4*)(k1 + 16*hi + 8);
        v8f zc = {};
        v8f sv0 = wmma_bf16(qf, kf0, zc);
        v8f sv1 = wmma_bf16(qf, kf1, zc);
        float x0[8], x1[8];
#pragma unroll
        for (int j = 0; j < 8; ++j) {
            int sr = srow0 + j + 8*hi;
            x0[j] = sv0[j]*scale + brow[sr*256 + t0 + lo];
            x1[j] = sv1[j]*scale + brow[sr*256 + t0 + 16 + lo];
        }
        // online softmax update (row reductions across the 16-lane half-groups)
#pragma unroll
        for (int j = 0; j < 8; ++j) {
            float rm = fmaxf(x0[j], x1[j]);
#pragma unroll
            for (int msk = 1; msk < 16; msk <<= 1) rm = fmaxf(rm, __shfl_xor(rm, msk));
            float mnew = fmaxf(mrow[j], rm);
            float alpha = __expf(mrow[j] - mnew);
            mrow[j] = mnew;
            float p0 = __expf(x0[j] - mnew);
            float p1 = __expf(x1[j] - mnew);
            float rs = p0 + p1;
#pragma unroll
            for (int msk = 1; msk < 16; msk <<= 1) rs += __shfl_xor(rs, msk);
            lrow[j] = lrow[j]*alpha + rs;
            o0[j] *= alpha;
            o1[j] *= alpha;
            int rl = j + 8*hi;                 // D-layout -> LDS bounce for A-layout reload
            Pscr[wave][rl*32 + lo]      = f2bf(p0);
            Pscr[wave][rl*32 + 16 + lo] = f2bf(p1);
        }
        asm volatile("s_wait_dscnt 0x0" ::: "memory");
        Frag16 pf;
        const unsigned short* prow = &Pscr[wave][lo*32];
        pf.q[0] = *(const uint4*)(prow + 16*hi);
        pf.q[1] = *(const uint4*)(prow + 16*hi + 8);
        // V fragments: B cols c=lo and c=lo+16, K-range t = t0 + 16*hi (contiguous in Vt)
        Frag16 vf0, vf1;
        const unsigned short* v0 = Vt + lo*SEQ + t0 + 16*hi;
        const unsigned short* v1 = Vt + (lo+16)*SEQ + t0 + 16*hi;
        vf0.q[0] = *(const uint4*)(v0); vf0.q[1] = *(const uint4*)(v0 + 8);
        vf1.q[0] = *(const uint4*)(v1); vf1.q[1] = *(const uint4*)(v1 + 8);
        o0 = wmma_bf16(pf, vf0, o0);
        o1 = wmma_bf16(pf, vf1, o1);
    }
    // epilogue: normalize, gate, store go[s][l][h*32+c] bf16
#pragma unroll
    for (int j = 0; j < 8; ++j) {
        int sr = srow0 + j + 8*hi;
        float inv = 1.f / lrow[j];
        size_t gb = ((size_t)sr*LLEN + l)*CM + h*CH;
        float g0 = bf2f(gate[gb + lo]);
        float g1 = bf2f(gate[gb + 16 + lo]);
        go[gb + lo]      = f2bf(o0[j]*inv*g0);
        go[gb + 16 + lo] = f2bf(o1[j]*inv*g1);
    }
}

extern "C" void kernel_launch(void* const* d_in, const int* in_sizes, int n_in,
                              void* d_out, int out_size, void* d_ws, size_t ws_size,
                              hipStream_t stream) {
    (void)in_sizes; (void)n_in; (void)out_size; (void)ws_size;
    const float* m   = (const float*)d_in[0];
    const float* z   = (const float*)d_in[1];
    const float* g_m = (const float*)d_in[2];
    const float* b_m = (const float*)d_in[3];
    const float* g_z = (const float*)d_in[4];
    const float* b_z = (const float*)d_in[5];
    const float* Wq  = (const float*)d_in[6];
    const float* Wk  = (const float*)d_in[7];
    const float* Wv  = (const float*)d_in[8];
    const float* Wb  = (const float*)d_in[9];
    const float* Wg  = (const float*)d_in[10];
    const float* bg  = (const float*)d_in[11];
    const float* Wo  = (const float*)d_in[12];
    const float* bo  = (const float*)d_in[13];
    float* out = (float*)d_out;

    char* ws = (char*)d_ws;
    size_t off = 0;
    auto alloc = [&](size_t bytes) -> char* {
        char* p = ws + off; off += (bytes + 255) & ~(size_t)255; return p;
    };
    unsigned short* m_ln = (unsigned short*)alloc((size_t)ROWS*CM*2);   // 32 MB
    unsigned short* WtQ  = (unsigned short*)alloc(256*256*2);
    unsigned short* WtK  = (unsigned short*)alloc(256*256*2);
    unsigned short* WtV  = (unsigned short*)alloc(256*256*2);
    unsigned short* WtG  = (unsigned short*)alloc(256*256*2);
    unsigned short* WtO  = (unsigned short*)alloc(256*256*2);
    unsigned short* qp   = (unsigned short*)alloc((size_t)ROWS*CM*2);   // [l][h][s][c]
    unsigned short* kp   = (unsigned short*)alloc((size_t)ROWS*CM*2);
    unsigned short* vp   = (unsigned short*)alloc((size_t)ROWS*CM*2);
    unsigned short* gatep= (unsigned short*)alloc((size_t)ROWS*CM*2);
    float*          biasw= (float*)alloc((size_t)NH*ROWS*4);            // 2 MB
    unsigned short* gop  = (unsigned short*)alloc((size_t)ROWS*CM*2);

    dim3 tb(16, 16);
    dim3 tg(16, 16);
    ln_m_kernel<<<ROWS/8, 256, 0, stream>>>(m, g_m, b_m, m_ln);
    pair_bias_kernel<<<ROWS/8, 256, 0, stream>>>(z, g_z, b_z, Wb, biasw);
    wt_convert_kernel<<<tg, tb, 0, stream>>>(Wq, WtQ, 256, 256);
    wt_convert_kernel<<<tg, tb, 0, stream>>>(Wk, WtK, 256, 256);
    wt_convert_kernel<<<tg, tb, 0, stream>>>(Wv, WtV, 256, 256);
    wt_convert_kernel<<<tg, tb, 0, stream>>>(Wg, WtG, 256, 256);
    wt_convert_kernel<<<tg, tb, 0, stream>>>(Wo, WtO, 256, 256);

    int gemm_blocks = (ROWS/16) * (256/16) / 8;   // 8192
    proj_fused_kernel<<<gemm_blocks, 256, 0, stream>>>(m_ln, WtQ, WtK, WtV, WtG, bg,
                                                       qp, kp, vp, gatep);

    attn_kernel<<<LLEN*NH, 512, 0, stream>>>(qp, kp, vp, biasw, gatep, gop);

    gemm_out_kernel<<<gemm_blocks, 256, 0, stream>>>(gop, WtO, bo, out);
}
